// Qwen3MoeSparseMoeBlock_46926812676665
// MI455X (gfx1250) — matrix operand order repaired and checked by reading, not compiled, not run on hardware
//
#include <hip/hip_runtime.h>
#include <hip/hip_bf16.h>

// ---------------------------------------------------------------------------
// Qwen3 MoE sparse block for MI455X (gfx1250, wave32, WMMA)
// B=4 S=2048 D=1024 E=64 F=768 TOP_K=8  -> T=8192 tokens
//
// kernel 1: router (fp32 VALU)   logits -> top8 -> renorm -> per-expert lists
// kernel 2: grouped fused MoE    h = silu(x Wg) * (x Wu); out += (h Wd)*w
//           bf16 v_wmma_f32_16x16x32_bf16, h kept in LDS (never leaves WGP)
// ---------------------------------------------------------------------------

#define T_TOK   8192
#define DIM     1024
#define NEXP    64
#define FDIM    768
#define TOPK    8
#define BM      64            // token rows per block
#define MAX_TILES (T_TOK / BM) // 128

typedef __attribute__((ext_vector_type(16))) __bf16 bf16x16;
typedef __attribute__((ext_vector_type(8)))  float  f32x8;

union FragU { bf16x16 v; uint4 u[2]; };

// A-fragment (16x32 bf16) from a row-major (K-contiguous) LDS tile.
// ISA layout: lane m=lane&15; VGPR0..3 hold K = (lane<16?0:8)+0..7,
//             VGPR4..7 hold K = 16 + (lane<16?0:8)+0..7
__device__ __forceinline__ bf16x16 lds_a_frag(const __bf16* sh, int ld,
                                              int row0, int k0, int lane) {
  const int m  = lane & 15;
  const int hi = lane >> 4;
  const __bf16* p = sh + (row0 + m) * ld + k0 + hi * 8;
  FragU f;
  f.u[0] = *reinterpret_cast<const uint4*>(p);
  f.u[1] = *reinterpret_cast<const uint4*>(p + 16);
  return f.v;
}

// B-fragment (32x16 bf16) straight from a row-major fp32 weight matrix
// (row = K, stride ld, N contiguous), converting fp32 -> bf16 in VALU.
// ISA layout: lane holds N = n0+(lane&15); elems 0..7 K = k0+(lane<16?0:8)+0..7,
//             elems 8..15 K = k0+16+(lane<16?0:8)+0..7.
__device__ __forceinline__ bf16x16 glb_b_frag(const float* __restrict__ W,
                                              int ld, int k0, int n0, int lane) {
  const float* p = W + (size_t)(k0 + (lane >> 4) * 8) * ld + n0 + (lane & 15);
  bf16x16 r;
#pragma unroll
  for (int s = 0; s < 8; ++s) {
    r[s]     = (__bf16)p[(size_t)s * ld];
    r[8 + s] = (__bf16)p[(size_t)(16 + s) * ld];
  }
  return r;
}

__device__ __forceinline__ f32x8 wmma_bf16(bf16x16 a, bf16x16 b, f32x8 c) {
  return __builtin_amdgcn_wmma_f32_16x16x32_bf16(
      /*neg_a=*/false, a, /*neg_b=*/false, b,
      /*c_mod=*/(short)0, c, /*reuse_a=*/false, /*reuse_b=*/false);
}

// ---------------------------------------------------------------------------
// Router: one wave per token.  fp32 logits (64), top-8 by logit, softmax over
// the selected 8 (identical to softmax-then-topk-then-renorm), scatter token
// into per-expert list with its combine weight.
// ---------------------------------------------------------------------------
__global__ __launch_bounds__(256) void router_kernel(
    const float* __restrict__ x, const float* __restrict__ gw,
    int* __restrict__ counts, int* __restrict__ list, float* __restrict__ wlist) {
  const int lane = threadIdx.x & 31;
  const int wv   = threadIdx.x >> 5;
  const int t    = blockIdx.x * 8 + wv;
  if (t >= T_TOK) return;

  // x row cached in registers: lane holds columns lane, lane+32, ...
  float xr[32];
#pragma unroll
  for (int i = 0; i < 32; ++i) xr[i] = x[(size_t)t * DIM + i * 32 + lane];

  float v0 = -1e30f, v1 = -1e30f;   // logits for experts lane and lane+32
  for (int e = 0; e < NEXP; ++e) {
    const float* w = gw + (size_t)e * DIM;
    float acc = 0.f;
#pragma unroll
    for (int i = 0; i < 32; ++i) acc += xr[i] * w[i * 32 + lane];
#pragma unroll
    for (int off = 16; off; off >>= 1) acc += __shfl_xor(acc, off, 32);
    if (e == lane)      v0 = acc;
    if (e == lane + 32) v1 = acc;
  }

  // iterative wave-wide argmax (ties -> lower index, matching lax.top_k)
  float lmax = 0.f, my_l = 0.f;
  int   my_e = 0;
#pragma unroll
  for (int k = 0; k < TOPK; ++k) {
    float best; int bidx;
    if (v0 >= v1) { best = v0; bidx = lane; } else { best = v1; bidx = lane + 32; }
#pragma unroll
    for (int off = 16; off; off >>= 1) {
      float ov = __shfl_xor(best, off, 32);
      int   oi = __shfl_xor(bidx, off, 32);
      if (ov > best || (ov == best && oi < bidx)) { best = ov; bidx = oi; }
    }
    if (k == 0) lmax = best;
    if (lane == k) { my_l = best; my_e = bidx; }
    if (bidx == lane)           v0 = -1e30f;
    else if (bidx == lane + 32) v1 = -1e30f;
  }

  float term = (lane < TOPK) ? __expf(my_l - lmax) : 0.f;
  float s = term;
#pragma unroll
  for (int off = 16; off; off >>= 1) s += __shfl_xor(s, off, 32);

  if (lane < TOPK) {
    const float wgt = term / s;
    const int pos = atomicAdd(counts + my_e, 1);
    list[(size_t)my_e * T_TOK + pos]  = t;
    wlist[(size_t)my_e * T_TOK + pos] = wgt;
  }
}

// ---------------------------------------------------------------------------
// Grouped fused expert kernel.  grid = (MAX_TILES, NEXP), 256 thr = 8 waves.
// LDS: full 64x1024 bf16 x-tile + 64x768 bf16 h-tile (~227 KB of the WGP's
// 320 KB) so the SwiGLU intermediate never touches HBM.
// ---------------------------------------------------------------------------
__global__ __launch_bounds__(256) void moe_expert_kernel(
    const float* __restrict__ x,
    const float* __restrict__ gate_proj,
    const float* __restrict__ up_proj,
    const float* __restrict__ down_proj,
    const int*   __restrict__ counts,
    const int*   __restrict__ list,
    const float* __restrict__ wlist,
    float*       __restrict__ out) {
  constexpr int LDX = DIM + 8;    // +16B pad -> LDS stride = 4 banks
  constexpr int LDH = FDIM + 8;

  __shared__ alignas(16) __bf16 sh_x[BM * LDX];   // 129 KB
  __shared__ alignas(16) __bf16 sh_h[BM * LDH];   //  97 KB
  __shared__ int   sh_tok[BM];
  __shared__ float sh_w[BM];

  const int e        = blockIdx.y;
  const int tile     = blockIdx.x;
  const int cnt      = counts[e];
  const int row_base = tile * BM;
  if (row_base >= cnt) return;                    // empty tile, whole block
  const int nvalid = min(BM, cnt - row_base);

  const int tid  = threadIdx.x;
  const int lane = tid & 31;
  const int wv   = tid >> 5;

  if (tid < BM) {
    const bool v = tid < nvalid;
    sh_tok[tid] = v ? list[(size_t)e * T_TOK + row_base + tid] : 0;
    sh_w[tid]   = v ? wlist[(size_t)e * T_TOK + row_base + tid] : 0.f;
  }
  __syncthreads();

  // ---- stage gathered x tile (BM x DIM) as bf16, fp32->bf16 on the fly ----
  for (int i = tid; i < BM * (DIM / 4); i += 256) {
    const int r  = i >> 8;            // / (DIM/4)
    const int c4 = i & 255;
    const float4 f =
        reinterpret_cast<const float4*>(x + (size_t)sh_tok[r] * DIM)[c4];
    __bf16* d = &sh_x[r * LDX + c4 * 4];
    d[0] = (__bf16)f.x; d[1] = (__bf16)f.y; d[2] = (__bf16)f.z; d[3] = (__bf16)f.w;
  }
  __syncthreads();

  const float* Wg = gate_proj + (size_t)e * DIM * FDIM;
  const float* Wu = up_proj   + (size_t)e * DIM * FDIM;
  const float* Wd = down_proj + (size_t)e * FDIM * DIM;

  const int nn = lane & 15;
  const int hi = lane >> 4;

  // =========================== GEMM1 (SwiGLU) ==============================
  // wave wv owns F-columns [wv*96, wv*96+96), processed as 2 chunks of 48.
  for (int c = 0; c < 2; ++c) {
    const int n_base = wv * 96 + c * 48;
    f32x8 acc[4][3];

    // ---- up projection: u = x @ Wu  -> stash bf16 into sh_h ----
#pragma unroll
    for (int mt = 0; mt < 4; ++mt)
#pragma unroll
      for (int nt = 0; nt < 3; ++nt) acc[mt][nt] = (f32x8){};

    for (int k0 = 0; k0 < DIM; k0 += 32) {
      bf16x16 a[4];
#pragma unroll
      for (int mt = 0; mt < 4; ++mt)
        a[mt] = lds_a_frag(sh_x, LDX, mt * 16, k0, lane);
#pragma unroll
      for (int nt = 0; nt < 3; ++nt) {
        const bf16x16 b = glb_b_frag(Wu, FDIM, k0, n_base + nt * 16, lane);
#pragma unroll
        for (int mt = 0; mt < 4; ++mt) acc[mt][nt] = wmma_bf16(a[mt], b, acc[mt][nt]);
      }
    }
#pragma unroll
    for (int mt = 0; mt < 4; ++mt)
#pragma unroll
      for (int nt = 0; nt < 3; ++nt)
#pragma unroll
        for (int r = 0; r < 8; ++r)
          sh_h[(mt * 16 + r + 8 * hi) * LDH + n_base + nt * 16 + nn] =
              (__bf16)acc[mt][nt][r];

    // ---- gate projection: g = x @ Wg ----
#pragma unroll
    for (int mt = 0; mt < 4; ++mt)
#pragma unroll
      for (int nt = 0; nt < 3; ++nt) acc[mt][nt] = (f32x8){};

    for (int k0 = 0; k0 < DIM; k0 += 32) {
      bf16x16 a[4];
#pragma unroll
      for (int mt = 0; mt < 4; ++mt)
        a[mt] = lds_a_frag(sh_x, LDX, mt * 16, k0, lane);
#pragma unroll
      for (int nt = 0; nt < 3; ++nt) {
        const bf16x16 b = glb_b_frag(Wg, FDIM, k0, n_base + nt * 16, lane);
#pragma unroll
        for (int mt = 0; mt < 4; ++mt) acc[mt][nt] = wmma_bf16(a[mt], b, acc[mt][nt]);
      }
    }

    // ---- h = silu(g) * u, overwrite stash in place ----
#pragma unroll
    for (int mt = 0; mt < 4; ++mt)
#pragma unroll
      for (int nt = 0; nt < 3; ++nt)
#pragma unroll
        for (int r = 0; r < 8; ++r) {
          const int idx = (mt * 16 + r + 8 * hi) * LDH + n_base + nt * 16 + nn;
          const float g = acc[mt][nt][r];
          const float u = (float)sh_h[idx];
          const float h = g / (1.f + __expf(-g)) * u;
          sh_h[idx] = (__bf16)h;
        }
  }
  __syncthreads();   // h tile complete & visible to all waves

  // =========================== GEMM2 (down proj) ===========================
  // wave wv owns D-columns [wv*128, wv*128+128), 4 sub-chunks of 32.
  for (int sc = 0; sc < 4; ++sc) {
    const int n0 = wv * 128 + sc * 32;
    f32x8 acc[4][2];
#pragma unroll
    for (int mt = 0; mt < 4; ++mt)
#pragma unroll
      for (int nt = 0; nt < 2; ++nt) acc[mt][nt] = (f32x8){};

    for (int k0 = 0; k0 < FDIM; k0 += 32) {
      bf16x16 a[4];
#pragma unroll
      for (int mt = 0; mt < 4; ++mt)
        a[mt] = lds_a_frag(sh_h, LDH, mt * 16, k0, lane);
#pragma unroll
      for (int nt = 0; nt < 2; ++nt) {
        const bf16x16 b = glb_b_frag(Wd, DIM, k0, n0 + nt * 16, lane);
#pragma unroll
        for (int mt = 0; mt < 4; ++mt) acc[mt][nt] = wmma_bf16(a[mt], b, acc[mt][nt]);
      }
    }

    // scale by per-token combine weight, scatter-accumulate into out
#pragma unroll
    for (int mt = 0; mt < 4; ++mt)
#pragma unroll
      for (int nt = 0; nt < 2; ++nt)
#pragma unroll
        for (int r = 0; r < 8; ++r) {
          const int row = mt * 16 + r + 8 * hi;
          if (row < nvalid) {
            const int col = n0 + nt * 16 + nn;
            atomicAdd(out + (size_t)sh_tok[row] * DIM + col,
                      acc[mt][nt][r] * sh_w[row]);
          }
        }
  }
}

// ---------------------------------------------------------------------------
__global__ void zero_kernel(unsigned int* __restrict__ p, size_t n) {
  size_t i = (size_t)blockIdx.x * blockDim.x + threadIdx.x;
  const size_t stride = (size_t)gridDim.x * blockDim.x;
  for (; i < n; i += stride) p[i] = 0u;
}

extern "C" void kernel_launch(void* const* d_in, const int* in_sizes, int n_in,
                              void* d_out, int out_size, void* d_ws, size_t ws_size,
                              hipStream_t stream) {
  (void)in_sizes; (void)n_in; (void)out_size; (void)ws_size;
  const float* x  = (const float*)d_in[0];   // [4,2048,1024]
  const float* gw = (const float*)d_in[1];   // [64,1024]
  const float* wg = (const float*)d_in[2];   // [64,1024,768]
  const float* wu = (const float*)d_in[3];   // [64,1024,768]
  const float* wd = (const float*)d_in[4];   // [64,768,1024]
  float* out = (float*)d_out;                // [4,2048,1024]

  char* ws = (char*)d_ws;
  int*   counts = (int*)ws;                                        // 64 ints
  int*   list   = (int*)(ws + 1024);                               // 2 MB
  float* wlist  = (float*)(ws + 1024 + (size_t)NEXP * T_TOK * 4);  // 2 MB

  zero_kernel<<<2048, 256, 0, stream>>>((unsigned int*)out, (size_t)T_TOK * DIM);
  zero_kernel<<<1, 64, 0, stream>>>((unsigned int*)counts, (size_t)NEXP);

  router_kernel<<<T_TOK / 8, 256, 0, stream>>>(x, gw, counts, list, wlist);

  dim3 grid(MAX_TILES, NEXP);
  moe_expert_kernel<<<grid, 256, 0, stream>>>(x, wg, wu, wd, counts, list, wlist, out);
}
